// LSTMAttentionModel_20615843021545
// MI455X (gfx1250) — matrix operand order repaired and checked by reading, not compiled, not run on hardware
//
#include <hip/hip_runtime.h>
#include <hip/hip_bf16.h>

typedef __attribute__((ext_vector_type(16))) _Float16 v16h;
typedef __attribute__((ext_vector_type(8)))  _Float16 v8h;
typedef __attribute__((ext_vector_type(8)))  float    v8f;

union Frag16 { v16h v; v8h h[2]; };

// ---- model dims ----
#define S_LEN 128
#define B_SZ  256
#define H_DIM 512
#define E_DIM 256
#define V_SZ  50000
#define H4    2048

// ---- GEMM tiling ----
#define BM 128
#define BN 128
#define BK 32
#define LDSP 40   // padded LDS row stride in halves (80B; 16B-aligned, conflict-avoiding)

// ---- CDNA5 async global->LDS copy (ASYNCcnt-tracked, bypasses VGPRs) ----
__device__ __forceinline__ void async_ld_b128(unsigned lds_off, const void* gptr) {
  asm volatile("global_load_async_to_lds_b128 %0, %1, off"
               :: "v"(lds_off), "v"(gptr) : "memory");
}
__device__ __forceinline__ void wait_async0() {
  asm volatile("s_wait_asynccnt 0" ::: "memory");
}
__device__ __forceinline__ void wait_async2() {
  asm volatile("s_wait_asynccnt 2" ::: "memory");
}
__device__ __forceinline__ void wait_async4() {
  asm volatile("s_wait_asynccnt 4" ::: "memory");
}

// ============================================================================
// Generic WMMA GEMM:  C = act( accum?C:0 + A(MxK,f16) * B^T + bias )
//   BTRANS=1: Bm is weight-style [N,K] row-major (C = A @ Bm^T)
//   BTRANS=0: Bm is [K,N] row-major        (C = A @ Bm)
// Double-buffered LDS; tiles staged with global_load_async_to_lds_b128 and
// consumed by v_wmma_f32_16x16x32_f16. grid=(ceil(N/128),ceil(M/128),batch),
// block = 256 threads (8 waves, wave32).
// ============================================================================
template<int ACT, bool BIAS, bool ACCUM, bool BTRANS>
__global__ __launch_bounds__(256) void gemm_wmma(
    const _Float16* __restrict__ A, const _Float16* __restrict__ Bm,
    const float* __restrict__ bias, float* __restrict__ C32,
    _Float16* __restrict__ C16,
    int M, int N, int K, int lda, int ldb, int ldc,
    long long sA, long long sB, long long sC)
{
  __shared__ __align__(16) _Float16 As[2 * BM * LDSP];
  __shared__ __align__(16) _Float16 Bs[2 * BN * LDSP];

  const int bz = blockIdx.z;
  A  += (long long)bz * sA;
  Bm += (long long)bz * sB;

  const int m0   = blockIdx.y * BM;
  const int n0   = blockIdx.x * BN;
  const int tid  = threadIdx.x;
  const int lane = tid & 31;
  const int wave = tid >> 5;       // 8 waves
  const int wm   = wave >> 1;      // 0..3 -> 32-row slab
  const int wn   = wave & 1;       // 0..1 -> 64-col slab
  const int lr   = lane & 15;      // row/col within a 16x16 tile
  const int lh   = lane >> 4;      // half-wave select (ISA A/B layout)

  const bool fullTile = (m0 + BM <= M) && (n0 + BN <= N);
  const int  nk = K / BK;          // K is a multiple of BK in all uses

  v8f acc[2][4];
  #pragma unroll
  for (int i = 0; i < 2; ++i)
    #pragma unroll
    for (int j = 0; j < 4; ++j)
      #pragma unroll
      for (int r = 0; r < 8; ++r) acc[i][j][r] = 0.f;

  // ---- tile stager: async HBM->LDS for A (and B in NT); sync transpose for NN B ----
  auto stage = [&](int it, int buf) {
    const int k0 = it * BK;
    _Float16* Ab = As + buf * (BM * LDSP);
    _Float16* Bb = Bs + buf * (BN * LDSP);
    // A tile (BM x BK): 2 async b128 per thread
    #pragma unroll
    for (int v = tid; v < (BM * BK) / 8; v += 256) {
      int m  = v >> 2;
      int kv = (v & 3) * 8;
      int gm = m0 + m;
      if (gm < M) {
        async_ld_b128((unsigned)(size_t)&Ab[m * LDSP + kv],
                      A + (long long)gm * lda + k0 + kv);
      } else {
        v8h z;
        #pragma unroll
        for (int j = 0; j < 8; ++j) z[j] = (_Float16)0.f;
        *(v8h*)&Ab[m * LDSP + kv] = z;
      }
    }
    if (BTRANS) {
      // B tile as Bs[n][k]: direct row-major copy, 2 async b128 per thread
      #pragma unroll
      for (int v = tid; v < (BN * BK) / 8; v += 256) {
        int n  = v >> 2;
        int kv = (v & 3) * 8;
        int gn = n0 + n;
        if (gn < N) {
          async_ld_b128((unsigned)(size_t)&Bb[n * LDSP + kv],
                        Bm + (long long)gn * ldb + k0 + kv);
        } else {
          v8h z;
          #pragma unroll
          for (int j = 0; j < 8; ++j) z[j] = (_Float16)0.f;
          *(v8h*)&Bb[n * LDSP + kv] = z;
        }
      }
    } else {
      // Bm is [K,N]: transpose-stage into Bs[n][k] (sync; async can't transpose)
      #pragma unroll
      for (int v = tid; v < (BK * BN) / 8; v += 256) {
        int k = v >> 4;
        int n = (v & 15) * 8;
        v8h val;
        #pragma unroll
        for (int j = 0; j < 8; ++j) val[j] = (_Float16)0.f;
        if (n0 + n < N) val = *(const v8h*)(Bm + (long long)(k0 + k) * ldb + n0 + n);
        #pragma unroll
        for (int j = 0; j < 8; ++j) Bb[(n + j) * LDSP + k] = val[j];
      }
    }
  };

  stage(0, 0);  // prologue: fill buffer 0

  for (int it = 0; it < nk; ++it) {
    const int  cur      = it & 1;
    const bool has_next = (it + 1 < nk);
    if (has_next) stage(it + 1, cur ^ 1);   // overlap next tile copy with compute

    // ASYNCcnt is per-wave and in-order: waiting <= (ops issued for next tile)
    // guarantees the current tile's copies have landed in LDS.
    if (has_next && fullTile) {
      if (BTRANS) wait_async4(); else wait_async2();
    } else {
      wait_async0();
    }
    __syncthreads();

    const _Float16* Ab = As + cur * (BM * LDSP);
    const _Float16* Bb = Bs + cur * (BN * LDSP);

    // ---- fragments per ISA 7.12.2 layouts + WMMA ----
    Frag16 afr[2];
    #pragma unroll
    for (int ti = 0; ti < 2; ++ti) {
      int m = wm * 32 + ti * 16 + lr;
      // A 16x32 f16: halves 0-7 -> K = lh*8+0..7 ; halves 8-15 -> K = 16+lh*8+0..7
      afr[ti].h[0] = *(const v8h*)&Ab[m * LDSP + lh * 8];
      afr[ti].h[1] = *(const v8h*)&Ab[m * LDSP + 16 + lh * 8];
    }
    #pragma unroll
    for (int tj = 0; tj < 4; ++tj) {
      Frag16 bfr;
      int n = wn * 64 + tj * 16 + lr;
      // B 32x16 f16: halves 0-7 -> K = lh*8+0..7 ; halves 8-15 -> K = 16+lh*8+0..7
      bfr.h[0] = *(const v8h*)&Bb[n * LDSP + lh * 8];
      bfr.h[1] = *(const v8h*)&Bb[n * LDSP + 16 + lh * 8];
      #pragma unroll
      for (int ti = 0; ti < 2; ++ti) {
        acc[ti][tj] = __builtin_amdgcn_wmma_f32_16x16x32_f16(
            false, afr[ti].v, false, bfr.v, (short)0, acc[ti][tj], false, false);
      }
    }
    __syncthreads();  // compute done before this buffer is overwritten at it+2
  }

  // ---- epilogue ----
  float*    c32 = C32 ? C32 + (long long)bz * sC : nullptr;
  _Float16* c16 = C16 ? C16 + (long long)bz * sC : nullptr;
  #pragma unroll
  for (int ti = 0; ti < 2; ++ti) {
    #pragma unroll
    for (int tj = 0; tj < 4; ++tj) {
      int col = n0 + wn * 64 + tj * 16 + lr;
      #pragma unroll
      for (int r = 0; r < 8; ++r) {
        int row = m0 + wm * 32 + ti * 16 + lh * 8 + r;  // C/D VGPR layout
        if (row < M && col < N) {
          float val = acc[ti][tj][r];
          if (ACCUM) val += c32[(long long)row * ldc + col];
          if (BIAS)  val += bias[col];
          if (ACT == 1) val = tanhf(val);
          if (ACT == 2) val = fmaxf(val, 0.f);
          if (c32) c32[(long long)row * ldc + col] = val;
          if (c16) c16[(long long)row * ldc + col] = (_Float16)val;
        }
      }
    }
  }
}

// ============================================================================
// Elementwise / helper kernels
// ============================================================================
__global__ void f32_to_f16_kernel(const float* __restrict__ src,
                                  _Float16* __restrict__ dst, int n) {
  int i = blockIdx.x * blockDim.x + threadIdx.x;
  if (i < n) dst[i] = (_Float16)src[i];
}

__global__ void zero_f32_kernel(float* __restrict__ p, int n) {
  int i = blockIdx.x * blockDim.x + threadIdx.x;
  if (i < n) p[i] = 0.f;
}

__global__ void zero_f16_kernel(_Float16* __restrict__ p, int n) {
  int i = blockIdx.x * blockDim.x + threadIdx.x;
  if (i < n) p[i] = (_Float16)0.f;
}

__global__ void embed_gather_kernel(const int* __restrict__ x,
                                    const float* __restrict__ emb,
                                    _Float16* __restrict__ X16, int total) {
  int i = blockIdx.x * blockDim.x + threadIdx.x;
  if (i >= total) return;
  int sb = i >> 8;   // / E_DIM (=256)
  int e  = i & 255;
  int tok = x[sb];
  X16[i] = (_Float16)emb[(long long)tok * E_DIM + e];
}

__device__ __forceinline__ float sigmf(float x) { return 1.f / (1.f + __expf(-x)); }

__global__ void lstm_gate_kernel(const float* __restrict__ Z, float* __restrict__ h,
                                 float* __restrict__ c, _Float16* __restrict__ h16,
                                 _Float16* __restrict__ out16,
                                 const int* __restrict__ lengths, int t) {
  int idx = blockIdx.x * blockDim.x + threadIdx.x;
  if (idx >= B_SZ * H_DIM) return;
  int b  = idx >> 9;    // / 512
  int hh = idx & 511;
  const float* z = Z + (long long)b * H4;      // torch gate order: i,f,g,o
  float ig = sigmf(z[hh]);
  float fg = sigmf(z[H_DIM + hh]);
  float gg = tanhf(z[2 * H_DIM + hh]);
  float og = sigmf(z[3 * H_DIM + hh]);
  float cn = fg * c[idx] + ig * gg;
  float hn = og * tanhf(cn);
  c[idx] = cn; h[idx] = hn; h16[idx] = (_Float16)hn;
  bool valid = t < lengths[b];   // == reference's pad_packed masking
  out16[((long long)b * S_LEN + t) * H_DIM + hh] = valid ? (_Float16)hn : (_Float16)0.f;
}

__global__ void softmax_kernel(const float* __restrict__ score,
                               _Float16* __restrict__ attn,
                               const int* __restrict__ lengths) {
  int row = blockIdx.x;       // b*S + s
  int b   = row >> 7;         // / 128
  int t   = threadIdx.x;      // 0..127
  const float* sr = score + (long long)row * S_LEN;
  int len = lengths[b];
  float xv = (t < len) ? sr[t] : -3.4e38f;
  __shared__ float red[S_LEN];
  red[t] = xv; __syncthreads();
  for (int off = 64; off > 0; off >>= 1) {
    if (t < off) red[t] = fmaxf(red[t], red[t + off]);
    __syncthreads();
  }
  float mx = red[0]; __syncthreads();
  float ev = (t < len) ? __expf(xv - mx) : 0.f;
  red[t] = ev; __syncthreads();
  for (int off = 64; off > 0; off >>= 1) {
    if (t < off) red[t] += red[t + off];
    __syncthreads();
  }
  attn[(long long)row * S_LEN + t] = (_Float16)(ev / red[0]);
}

__global__ void mean_pool_kernel(const float* __restrict__ w,
                                 _Float16* __restrict__ p16) {
  int b = blockIdx.x;
  for (int hh = threadIdx.x; hh < H_DIM; hh += blockDim.x) {
    float s = 0.f;
    for (int t = 0; t < S_LEN; ++t) s += w[((long long)b * S_LEN + t) * H_DIM + hh];
    p16[b * H_DIM + hh] = (_Float16)(s * (1.f / S_LEN));
  }
}

// ============================================================================
// Host side
// ============================================================================
static inline void* bump(char*& p, size_t bytes) {
  void* r = p; p += (bytes + 255) & ~(size_t)255; return r;
}

template<int ACT, bool BIAS, bool ACCUM, bool BTRANS>
static void run_gemm(hipStream_t st, const _Float16* A, const _Float16* Bw,
                     const float* bias, float* C32, _Float16* C16,
                     int M, int N, int K, int lda, int ldb, int ldc,
                     long long sA = 0, long long sB = 0, long long sC = 0,
                     int batch = 1) {
  dim3 grid((N + BN - 1) / BN, (M + BM - 1) / BM, batch);
  gemm_wmma<ACT, BIAS, ACCUM, BTRANS><<<grid, dim3(256), 0, st>>>(
      A, Bw, bias, C32, C16, M, N, K, lda, ldb, ldc, sA, sB, sC);
}

extern "C" void kernel_launch(void* const* d_in, const int* in_sizes, int n_in,
                              void* d_out, int out_size, void* d_ws, size_t ws_size,
                              hipStream_t stream) {
  const float* emb  = (const float*)d_in[0];
  const float* Wih  = (const float*)d_in[1];
  const float* Whh  = (const float*)d_in[2];
  const float* bih  = (const float*)d_in[3];
  const float* bhh  = (const float*)d_in[4];
  const float* Wq   = (const float*)d_in[5];
  const float* bq   = (const float*)d_in[6];
  const float* Wk   = (const float*)d_in[7];
  const float* bk   = (const float*)d_in[8];
  const float* Wv   = (const float*)d_in[9];
  const float* bv   = (const float*)d_in[10];
  const float* W1   = (const float*)d_in[11];
  const float* b1   = (const float*)d_in[12];
  const float* W2   = (const float*)d_in[13];
  const float* b2   = (const float*)d_in[14];
  const float* Wh2e = (const float*)d_in[15];
  const float* bh2e = (const float*)d_in[16];
  const int*   x    = (const int*)d_in[17];
  const int*   lens = (const int*)d_in[18];
  float* logits = (float*)d_out;   // [B, V] f32

  // ---- workspace layout (bump-allocated, with reuse) ----
  char* p = (char*)d_ws;
  _Float16* emb16  = (_Float16*)bump(p, (size_t)V_SZ * E_DIM * 2);
  _Float16* Wih16  = (_Float16*)bump(p, (size_t)H4 * E_DIM * 2);
  _Float16* Whh16  = (_Float16*)bump(p, (size_t)H4 * H_DIM * 2);
  _Float16* Wq16   = (_Float16*)bump(p, (size_t)H_DIM * H_DIM * 2);
  _Float16* Wk16   = (_Float16*)bump(p, (size_t)H_DIM * H_DIM * 2);
  _Float16* Wv16   = (_Float16*)bump(p, (size_t)H_DIM * H_DIM * 2);
  _Float16* W116   = (_Float16*)bump(p, (size_t)H_DIM * H_DIM * 2);
  _Float16* W216   = (_Float16*)bump(p, (size_t)H_DIM * H_DIM * 2);
  _Float16* Wh2e16 = (_Float16*)bump(p, (size_t)E_DIM * H_DIM * 2);
  _Float16* X16    = (_Float16*)bump(p, (size_t)S_LEN * B_SZ * E_DIM * 2);
  float*    Z      = (float*)bump(p, (size_t)B_SZ * H4 * 4);
  float*    hbuf   = (float*)bump(p, (size_t)B_SZ * H_DIM * 4);
  float*    cbuf   = (float*)bump(p, (size_t)B_SZ * H_DIM * 4);
  _Float16* h16    = (_Float16*)bump(p, (size_t)B_SZ * H_DIM * 2);
  _Float16* out16  = (_Float16*)bump(p, (size_t)B_SZ * S_LEN * H_DIM * 2);
  _Float16* q16    = (_Float16*)bump(p, (size_t)B_SZ * S_LEN * H_DIM * 2);
  _Float16* k16    = (_Float16*)bump(p, (size_t)B_SZ * S_LEN * H_DIM * 2);
  _Float16* v16    = (_Float16*)bump(p, (size_t)B_SZ * S_LEN * H_DIM * 2);
  float*    tmp    = (float*)bump(p, (size_t)B_SZ * S_LEN * H_DIM * 4);
  _Float16* pool16 = (_Float16*)bump(p, (size_t)B_SZ * H_DIM * 2);
  _Float16* feat16 = (_Float16*)bump(p, (size_t)B_SZ * E_DIM * 2);
  // reuse: sh16 <- q16 (free after q@W1), attn <- k16 (free after k@W2),
  //        score(f32) <- X16 (free after LSTM; same byte size), weighted <- tmp.
  _Float16* sh16     = q16;
  _Float16* attn16   = k16;
  float*    score    = (float*)X16;
  float*    weighted = tmp;

  auto cvt = [&](const float* s, _Float16* d, int n) {
    f32_to_f16_kernel<<<(n + 255) / 256, 256, 0, stream>>>(s, d, n);
  };

  // ---- weight conversion to f16 (done per launch: deterministic) ----
  cvt(emb,  emb16,  V_SZ * E_DIM);
  cvt(Wih,  Wih16,  H4 * E_DIM);
  cvt(Whh,  Whh16,  H4 * H_DIM);
  cvt(Wq,   Wq16,   H_DIM * H_DIM);
  cvt(Wk,   Wk16,   H_DIM * H_DIM);
  cvt(Wv,   Wv16,   H_DIM * H_DIM);
  cvt(W1,   W116,   H_DIM * H_DIM);
  cvt(W2,   W216,   H_DIM * H_DIM);
  cvt(Wh2e, Wh2e16, E_DIM * H_DIM);

  // ---- embedding gather (f16) ----
  {
    int total = S_LEN * B_SZ * E_DIM;
    embed_gather_kernel<<<(total + 255) / 256, 256, 0, stream>>>(x, emb, X16, total);
  }

  // ---- init recurrent state ----
  zero_f32_kernel<<<(B_SZ * H_DIM + 255) / 256, 256, 0, stream>>>(hbuf, B_SZ * H_DIM);
  zero_f32_kernel<<<(B_SZ * H_DIM + 255) / 256, 256, 0, stream>>>(cbuf, B_SZ * H_DIM);
  zero_f16_kernel<<<(B_SZ * H_DIM + 255) / 256, 256, 0, stream>>>(h16, B_SZ * H_DIM);

  // ---- LSTM recurrence (sequential critical path) ----
  for (int t = 0; t < S_LEN; ++t) {
    const _Float16* Xt = X16 + (long long)t * B_SZ * E_DIM;
    // Z = Xt @ Wih^T + bih
    run_gemm<0, true, false, true>(stream, Xt, Wih16, bih, Z, nullptr,
                                   B_SZ, H4, E_DIM, E_DIM, E_DIM, H4);
    // Z += h @ Whh^T + bhh
    run_gemm<0, true, true, true>(stream, h16, Whh16, bhh, Z, nullptr,
                                  B_SZ, H4, H_DIM, H_DIM, H_DIM, H4);
    lstm_gate_kernel<<<(B_SZ * H_DIM + 255) / 256, 256, 0, stream>>>(
        Z, hbuf, cbuf, h16, out16, lens, t);
  }

  // ---- attention (big parallel WMMA GEMMs) ----
  const int MS = B_SZ * S_LEN;  // 32768
  run_gemm<0, true, false, true>(stream, out16, Wq16, bq, nullptr, q16,
                                 MS, H_DIM, H_DIM, H_DIM, H_DIM, H_DIM);
  run_gemm<0, true, false, true>(stream, out16, Wk16, bk, nullptr, k16,
                                 MS, H_DIM, H_DIM, H_DIM, H_DIM, H_DIM);
  run_gemm<0, true, false, true>(stream, out16, Wv16, bv, nullptr, v16,
                                 MS, H_DIM, H_DIM, H_DIM, H_DIM, H_DIM);
  // tmp = q @ W1^T + b1
  run_gemm<0, true, false, true>(stream, q16, W116, b1, tmp, nullptr,
                                 MS, H_DIM, H_DIM, H_DIM, H_DIM, H_DIM);
  // sh16 = tanh(tmp + k @ W2^T + b2)   (sh16 aliases q16; q fully consumed above)
  run_gemm<1, true, true, true>(stream, k16, W216, b2, tmp, sh16,
                                MS, H_DIM, H_DIM, H_DIM, H_DIM, H_DIM);
  // score[b] = sh16[b] @ v16[b]^T   (batched NT, 128x128x512)
  run_gemm<0, false, false, true>(stream, sh16, v16, nullptr, score, nullptr,
                                  S_LEN, S_LEN, H_DIM, H_DIM, H_DIM, S_LEN,
                                  (long long)S_LEN * H_DIM, (long long)S_LEN * H_DIM,
                                  (long long)S_LEN * S_LEN, B_SZ);
  // masked softmax over t, write f16 attn (aliases k16; k fully consumed above)
  softmax_kernel<<<B_SZ * S_LEN, S_LEN, 0, stream>>>(score, attn16, lens);
  // weighted[b] = attn[b] @ v[b]   (batched NN, 128x512x128; aliases tmp)
  run_gemm<0, false, false, false>(stream, attn16, v16, nullptr, weighted, nullptr,
                                   S_LEN, H_DIM, S_LEN, S_LEN, H_DIM, H_DIM,
                                   (long long)S_LEN * S_LEN, (long long)S_LEN * H_DIM,
                                   (long long)S_LEN * H_DIM, B_SZ);
  mean_pool_kernel<<<B_SZ, 256, 0, stream>>>(weighted, pool16);
  // feat = relu(pooled @ Wh2e^T + bh2e)
  run_gemm<2, true, false, true>(stream, pool16, Wh2e16, bh2e, nullptr, feat16,
                                 B_SZ, E_DIM, H_DIM, H_DIM, H_DIM, E_DIM);
  // logits = feat @ emb^T  (N = 50000, bounds-guarded tiles)
  run_gemm<0, false, false, true>(stream, feat16, emb16, nullptr, logits, nullptr,
                                  B_SZ, V_SZ, E_DIM, E_DIM, E_DIM, V_SZ);
  (void)in_sizes; (void)n_in; (void)out_size; (void)ws_size;
}
